// Mamba_CT_Baseline_28716151341624
// MI455X (gfx1250) — matrix operand order repaired and checked by reading, not compiled, not run on hardware
//
#include <hip/hip_runtime.h>
#include <hip/hip_bf16.h>
#include <cstdint>
#include <cstddef>

// ---------------------------------------------------------------------------
// Types for CDNA5 WMMA (wave32): D(16x16,f32) = A(16x32,bf16) x B(32x16,bf16) + C
// ---------------------------------------------------------------------------
typedef __bf16          v16bf  __attribute__((ext_vector_type(16)));
typedef float           v8f    __attribute__((ext_vector_type(8)));
typedef unsigned short  u16x16 __attribute__((ext_vector_type(16)));

#define BN_EPS   1e-5f
#define BATCH    4
#define LSEQ     3240            // conv4 output: 10*18*18 (padded 1x1x1 conv)
#define BLM      (BATCH * LSEQ)  // 12960 GEMM rows = 32 * 405
#define DIN      256
#define DSTATE   16
#define DTRANK   8

// ---------------------------------------------------------------------------
// Helpers
// ---------------------------------------------------------------------------
__device__ __forceinline__ unsigned short f2bf(float f) {
  unsigned u = __builtin_bit_cast(unsigned, f);
  u += 0x7FFFu + ((u >> 16) & 1u);            // round-to-nearest-even
  return (unsigned short)(u >> 16);
}
__device__ __forceinline__ float siluf(float x)     { return x / (1.f + __expf(-x)); }
__device__ __forceinline__ float softplusf(float x) { return x > 20.f ? x : log1pf(__expf(x)); }

// A fragment: 16x32 bf16, LDS row-major [m][k].
// Lane l<16: M=l, halfs = K{0..7,16..23}; lane l>=16: M=l-16, halfs = K{8..15,24..31}.
// Both 8-half runs are 16B-contiguous -> lowers to ds_load_b128 pairs.
__device__ __forceinline__ v16bf frag_a(const unsigned short* s, int lane) {
  int m = lane & 15, g = lane >> 4;
  u16x16 u;
#pragma unroll
  for (int j = 0; j < 16; ++j) {
    int k = (j < 8) ? (g * 8 + j) : (16 + g * 8 + (j - 8));
    u[j] = s[m * 32 + k];
  }
  return __builtin_bit_cast(v16bf, u);
}

// B fragment: 32x16 bf16 from LDS staged as [n][k] rows (weight-major).
// Lane l<16: N=l, halfs = K{0..7,16..23}; lane l>=16: N=l-16, halfs = K{8..15,24..31}.
__device__ __forceinline__ v16bf frag_b(const unsigned short* s, int lane, int nbase) {
  int n = lane & 15, g = lane >> 4;
  u16x16 u;
#pragma unroll
  for (int j = 0; j < 16; ++j) u[j] = s[(nbase + n) * 32 + ((j < 8) ? (g * 8 + j) : (16 + g * 8 + (j - 8)))];
  return __builtin_bit_cast(v16bf, u);
}

// ---------------------------------------------------------------------------
// K1: conv1 (IC=2, K=54) direct, fused BN+ReLU.  x(4,2,32,64,64)->h1(4,64,16,32,32)
// ---------------------------------------------------------------------------
__global__ void conv1_kernel(const float* __restrict__ x, const float* __restrict__ w,
                             const float* __restrict__ cb, const float* __restrict__ bg,
                             const float* __restrict__ bb, float* __restrict__ out) {
  int idx = blockIdx.x * 256 + threadIdx.x;           // 4*64*16*32*32 total
  int ow = idx & 31, oh = (idx >> 5) & 31, od = (idx >> 10) & 15;
  int oc = (idx >> 14) & 63, b = idx >> 20;
  float acc = 0.f;
#pragma unroll
  for (int ic = 0; ic < 2; ++ic)
#pragma unroll
    for (int kd = 0; kd < 3; ++kd)
#pragma unroll
      for (int kh = 0; kh < 3; ++kh)
#pragma unroll
        for (int kw = 0; kw < 3; ++kw) {
          int id = od * 2 - 1 + kd, ih = oh * 2 - 1 + kh, iw = ow * 2 - 1 + kw;
          if ((unsigned)id < 32u && (unsigned)ih < 64u && (unsigned)iw < 64u)
            acc += x[(((size_t)b * 2 + ic) * 32 + id) * 4096 + ih * 64 + iw] *
                   w[((oc * 2 + ic) * 3 + kd) * 9 + kh * 3 + kw];
        }
  float s = bg[oc] * rsqrtf(1.f + BN_EPS);
  out[idx] = fmaxf((acc + cb[oc]) * s + bb[oc], 0.f);
}

// ---------------------------------------------------------------------------
// K2: 3x3x3 conv as implicit im2col GEMM with WMMA, fused BN+ReLU.
// Block = 128 threads (4 waves); tile M=32, N=64.
// Wave w: m-subtile = w>>1, n-half = w&1 -> 2 accumulators, 2 WMMA / K-step.
// K = IC*27 (64*27 = 1728, multiple of 32).
// ---------------------------------------------------------------------------
__global__ void __launch_bounds__(128)
conv3x3_wmma(const float* __restrict__ in, const float* __restrict__ w,
             const float* __restrict__ cb, const float* __restrict__ bg,
             const float* __restrict__ bb, float* __restrict__ out,
             int IC, int ID, int IH, int IW, int OC, int OD, int OH, int OW, int stride) {
  __shared__ unsigned short sA[32 * 32];
  __shared__ unsigned short sW[64 * 32];
  const int K = IC * 27;
  const int m0 = blockIdx.x * 32, n0 = blockIdx.y * 64;
  const int t = threadIdx.x, wave = t >> 5, lane = t & 31;
  const int m_sub = wave >> 1, n_half = wave & 1;
  v8f acc[2] = {};
  for (int k0 = 0; k0 < K; k0 += 32) {
    for (int i = t; i < 1024; i += 128) {                // im2col A tile 32x32
      int m = m0 + (i >> 5), kk = k0 + (i & 31);
      int ic = kk / 27, r = kk % 27;
      int kd = r / 9, kh = (r / 3) % 3, kw = r % 3;
      int ow = m % OW, tmp = m / OW;
      int oh = tmp % OH; tmp /= OH;
      int od = tmp % OD, b = tmp / OD;
      int id = od * stride - 1 + kd, ih = oh * stride - 1 + kh, iw = ow * stride - 1 + kw;
      float v = 0.f;
      if ((unsigned)id < (unsigned)ID && (unsigned)ih < (unsigned)IH && (unsigned)iw < (unsigned)IW)
        v = in[(((size_t)b * IC + ic) * ID + id) * (IH * IW) + ih * IW + iw];
      sA[i] = f2bf(v);
    }
    for (int i = t; i < 2048; i += 128) {                // weight tile 64x32
      int n = i >> 5, k = i & 31;
      sW[i] = f2bf(w[(size_t)(n0 + n) * K + k0 + k]);
    }
    __syncthreads();
    v16bf a = frag_a(sA + m_sub * 16 * 32, lane);
#pragma unroll
    for (int nt = 0; nt < 2; ++nt) {
      v16bf bfr = frag_b(sW, lane, n_half * 32 + nt * 16);
      acc[nt] = __builtin_amdgcn_wmma_f32_16x16x32_bf16(false, a, false, bfr, (short)0,
                                                        acc[nt], false, false);
    }
    __syncthreads();
  }
  const int nn = lane & 15, g = lane >> 4;
#pragma unroll
  for (int nt = 0; nt < 2; ++nt) {
    int oc = n0 + n_half * 32 + nt * 16 + nn;
    float s = bg[oc] * rsqrtf(1.f + BN_EPS);
    float cbv = cb[oc], bbv = bb[oc];
#pragma unroll
    for (int r = 0; r < 8; ++r) {
      int m = m0 + m_sub * 16 + g * 8 + r;
      int ow = m % OW, tmp = m / OW;
      int oh = tmp % OH; tmp /= OH;
      int od = tmp % OD, b = tmp / OD;
      out[((((size_t)b * OC + oc) * OD + od) * OH + oh) * OW + ow] =
          fmaxf((acc[nt][r] + cbv) * s + bbv, 0.f);
    }
  }
}

// ---------------------------------------------------------------------------
// K3: conv4 (1x1x1, padded) as GEMM over the padded (10,18,18) grid; writes
// the sequence directly in (B, L, 128) layout with fused BN+ReLU.
// ---------------------------------------------------------------------------
__global__ void __launch_bounds__(128)
conv4_wmma(const float* __restrict__ in /*(B,128,8,16,16)*/,
           const float* __restrict__ w /*(128,128)*/,
           const float* __restrict__ cb, const float* __restrict__ bg,
           const float* __restrict__ bb, float* __restrict__ out /*(B,L,128)*/) {
  __shared__ unsigned short sA[32 * 32];
  __shared__ unsigned short sW[64 * 32];
  const int m0 = blockIdx.x * 32, n0 = blockIdx.y * 64;
  const int t = threadIdx.x, wave = t >> 5, lane = t & 31;
  const int m_sub = wave >> 1, n_half = wave & 1;
  v8f acc[2] = {};
  for (int k0 = 0; k0 < 128; k0 += 32) {
    for (int i = t; i < 1024; i += 128) {
      int m = m0 + (i >> 5), k = k0 + (i & 31);
      int b = m / LSEQ, l = m % LSEQ;
      int dd = l / 324, rem = l % 324, hh = rem / 18, ww = rem % 18;
      int id = dd - 1, ih = hh - 1, iw = ww - 1;
      float v = 0.f;
      if ((unsigned)id < 8u && (unsigned)ih < 16u && (unsigned)iw < 16u)
        v = in[(((size_t)b * 128 + k) * 8 + id) * 256 + ih * 16 + iw];
      sA[i] = f2bf(v);
    }
    for (int i = t; i < 2048; i += 128) {
      int n = i >> 5, k = i & 31;
      sW[i] = f2bf(w[(size_t)(n0 + n) * 128 + k0 + k]);
    }
    __syncthreads();
    v16bf a = frag_a(sA + m_sub * 16 * 32, lane);
#pragma unroll
    for (int nt = 0; nt < 2; ++nt) {
      v16bf bfr = frag_b(sW, lane, n_half * 32 + nt * 16);
      acc[nt] = __builtin_amdgcn_wmma_f32_16x16x32_bf16(false, a, false, bfr, (short)0,
                                                        acc[nt], false, false);
    }
    __syncthreads();
  }
  const int nn = lane & 15, g = lane >> 4;
#pragma unroll
  for (int nt = 0; nt < 2; ++nt) {
    int oc = n0 + n_half * 32 + nt * 16 + nn;
    float s = bg[oc] * rsqrtf(1.f + BN_EPS);
    float cbv = cb[oc], bbv = bb[oc];
#pragma unroll
    for (int r = 0; r < 8; ++r) {
      int m = m0 + m_sub * 16 + g * 8 + r;
      out[(size_t)m * 128 + oc] = fmaxf((acc[nt][r] + cbv) * s + bbv, 0.f);
    }
  }
}

// ---------------------------------------------------------------------------
// K4: generic C(M,ldc) = A(M,K) x W(N,K)^T, bf16 WMMA, N masked (for N=40).
// M must be a multiple of 32 (BLM = 32*405).
// ---------------------------------------------------------------------------
__global__ void __launch_bounds__(128)
gemm_nt_wmma(const float* __restrict__ A, const float* __restrict__ W,
             float* __restrict__ C, int M, int N, int K, int ldc) {
  __shared__ unsigned short sA[32 * 32];
  __shared__ unsigned short sW[64 * 32];
  const int m0 = blockIdx.x * 32, n0 = blockIdx.y * 64;
  const int t = threadIdx.x, wave = t >> 5, lane = t & 31;
  const int m_sub = wave >> 1, n_half = wave & 1;
  v8f acc[2] = {};
  for (int k0 = 0; k0 < K; k0 += 32) {
    for (int i = t; i < 1024; i += 128) {
      int m = i >> 5, k = i & 31;
      sA[i] = f2bf(A[(size_t)(m0 + m) * K + k0 + k]);
    }
    for (int i = t; i < 2048; i += 128) {
      int n = n0 + (i >> 5), k = i & 31;
      sW[i] = f2bf(n < N ? W[(size_t)n * K + k0 + k] : 0.f);
    }
    __syncthreads();
    v16bf a = frag_a(sA + m_sub * 16 * 32, lane);
#pragma unroll
    for (int nt = 0; nt < 2; ++nt) {
      v16bf b = frag_b(sW, lane, n_half * 32 + nt * 16);
      acc[nt] = __builtin_amdgcn_wmma_f32_16x16x32_bf16(false, a, false, b, (short)0,
                                                        acc[nt], false, false);
    }
    __syncthreads();
  }
  const int nn = lane & 15, g = lane >> 4;
#pragma unroll
  for (int nt = 0; nt < 2; ++nt) {
    int gn = n0 + n_half * 32 + nt * 16 + nn;
    if (gn < N) {
#pragma unroll
      for (int r = 0; r < 8; ++r)
        C[(size_t)(m0 + m_sub * 16 + g * 8 + r) * ldc + gn] = acc[nt][r];
    }
  }
}

// ---------------------------------------------------------------------------
// K5: causal depthwise conv1d (width 4) + SiLU: xz[:, :256] -> u
// ---------------------------------------------------------------------------
__global__ void dwconv_silu_kernel(const float* __restrict__ xz, const float* __restrict__ w,
                                   const float* __restrict__ cb, float* __restrict__ u) {
  int idx = blockIdx.x * 256 + threadIdx.x;     // BL*256 threads
  int d = idx & 255, bl = idx >> 8;
  int b = bl / LSEQ, l = bl % LSEQ;
  float acc = cb[d];
#pragma unroll
  for (int j = 0; j < 4; ++j) {
    int ls = l - 3 + j;
    if (ls >= 0) acc += w[d * 4 + j] * xz[((size_t)b * LSEQ + ls) * 512 + d];
  }
  u[idx] = siluf(acc);
}

// ---------------------------------------------------------------------------
// K6: delta = softplus(dt @ dt_w^T + dt_b)  (K = 8, direct)
// ---------------------------------------------------------------------------
__global__ void delta_kernel(const float* __restrict__ xdbl, const float* __restrict__ dtw,
                             const float* __restrict__ dtb, float* __restrict__ delta) {
  int idx = blockIdx.x * 256 + threadIdx.x;     // BL*256 threads
  int d = idx & 255, bl = idx >> 8;
  float acc = dtb[d];
#pragma unroll
  for (int r = 0; r < DTRANK; ++r) acc += xdbl[(size_t)bl * 40 + r] * dtw[d * DTRANK + r];
  delta[idx] = softplusf(acc);
}

// ---------------------------------------------------------------------------
// K7: selective-scan. 16 lanes per channel d (one per state n), register-resident
// h, cross-lane reduction over states, fused u*D skip and silu(z) gate.
// grid = (DIN/16, BATCH), block = 256 (16 channels per block).
// ---------------------------------------------------------------------------
__global__ void __launch_bounds__(256)
scan_kernel(const float* __restrict__ xz, const float* __restrict__ u,
            const float* __restrict__ xdbl, const float* __restrict__ delta,
            const float* __restrict__ Alog, const float* __restrict__ Dp,
            float* __restrict__ y) {
  const int b = blockIdx.y;
  const int d = blockIdx.x * 16 + (threadIdx.x >> 4);
  const int n = threadIdx.x & 15;
  const float A = -__expf(Alog[d * DSTATE + n]);
  const float Dd = Dp[d];
  float h = 0.f;
  const size_t base = (size_t)b * LSEQ;
  for (int l = 0; l < LSEQ; ++l) {
    const size_t bl = base + l;
    if (l + 8 < LSEQ) {                        // -> global_prefetch_b8
      __builtin_prefetch(&delta[(bl + 8) * 256 + d]);
      __builtin_prefetch(&u[(bl + 8) * 256 + d]);
    }
    float dl = delta[bl * 256 + d];
    float uv = u[bl * 256 + d];
    float Bv = xdbl[bl * 40 + DTRANK + n];
    float Cv = xdbl[bl * 40 + DTRANK + DSTATE + n];
    float da = __expf(dl * A);
    h = da * h + dl * Bv * uv;
    float part = h * Cv;
    part += __shfl_xor(part, 1);
    part += __shfl_xor(part, 2);
    part += __shfl_xor(part, 4);
    part += __shfl_xor(part, 8);
    if (n == 0) {
      float zv = xz[bl * 512 + 256 + d];
      y[bl * 256 + d] = (part + uv * Dd) * siluf(zv);
    }
  }
}

// ---------------------------------------------------------------------------
// K8: mean-pool over L, fc1+ReLU, fc2. Single block of 256 threads.
// ---------------------------------------------------------------------------
__global__ void __launch_bounds__(256)
head_kernel(const float* __restrict__ seq, const float* __restrict__ fc1w,
            const float* __restrict__ fc1b, const float* __restrict__ fc2w,
            const float* __restrict__ fc2b, float* __restrict__ out) {
  __shared__ float pooled[BATCH * 128];
  __shared__ float hbuf[BATCH * 64];
  const int t = threadIdx.x;
  for (int i = t; i < BATCH * 128; i += 256) {
    int b = i >> 7, c = i & 127;
    float s = 0.f;
    for (int l = 0; l < LSEQ; ++l) s += seq[((size_t)b * LSEQ + l) * 128 + c];
    pooled[i] = s * (1.f / LSEQ);
  }
  __syncthreads();
  {
    int b = t >> 6, j = t & 63;
    float s = fc1b[j];
    for (int c = 0; c < 128; ++c) s += pooled[b * 128 + c] * fc1w[j * 128 + c];
    hbuf[t] = fmaxf(s, 0.f);
  }
  __syncthreads();
  if (t < BATCH * 2) {
    int b = t >> 1, o = t & 1;
    float s = fc2b[o];
    for (int j = 0; j < 64; ++j) s += hbuf[b * 64 + j] * fc2w[o * 64 + j];
    out[b * 2 + o] = s;
  }
}

// ---------------------------------------------------------------------------
// Workspace layout (floats). Conv scratch overlaps later Mamba scratch
// (disjoint lifetimes): conv h1/h2/h3 occupy [0, 9.44M) which sits inside
// [OFF_XZ, OFF_XDBL) = [0, 9.95M).
// ---------------------------------------------------------------------------
static constexpr size_t OFF_XZ    = 0;
static constexpr size_t SZ_XZ     = (size_t)BLM * 512;       // 6,635,520
static constexpr size_t OFF_U     = OFF_XZ + SZ_XZ;
static constexpr size_t SZ_CH     = (size_t)BLM * 256;       // 3,317,760
static constexpr size_t OFF_XDBL  = OFF_U + SZ_CH;           // 9,953,280
static constexpr size_t SZ_XDBL   = (size_t)BLM * 40;
static constexpr size_t OFF_DELTA = OFF_XDBL + SZ_XDBL;
static constexpr size_t OFF_Y     = OFF_DELTA + SZ_CH;
static constexpr size_t OFF_SEQA  = OFF_Y + SZ_CH;           // 17,107,200
static constexpr size_t OFF_SEQB  = OFF_SEQA + (size_t)BLM * 128;
static constexpr size_t OFF_H1    = 0;                       // 4,194,304 floats
static constexpr size_t OFF_H2    = OFF_H1 + 4194304;
static constexpr size_t OFF_H3    = OFF_H2 + 4194304;        // 1,048,576 floats

extern "C" void kernel_launch(void* const* d_in, const int* in_sizes, int n_in,
                              void* d_out, int out_size, void* d_ws, size_t ws_size,
                              hipStream_t stream) {
  (void)in_sizes; (void)n_in; (void)out_size; (void)ws_size;
  const float* x    = (const float*)d_in[0];
  const float* c1w  = (const float*)d_in[1];
  const float* c1b  = (const float*)d_in[2];
  const float* b1g  = (const float*)d_in[3];
  const float* b1b  = (const float*)d_in[4];
  const float* c2w  = (const float*)d_in[5];
  const float* c2b  = (const float*)d_in[6];
  const float* b2g  = (const float*)d_in[7];
  const float* b2b  = (const float*)d_in[8];
  const float* c3w  = (const float*)d_in[9];
  const float* c3b  = (const float*)d_in[10];
  const float* b3g  = (const float*)d_in[11];
  const float* b3b  = (const float*)d_in[12];
  const float* c4w  = (const float*)d_in[13];
  const float* c4b  = (const float*)d_in[14];
  const float* b4g  = (const float*)d_in[15];
  const float* b4b  = (const float*)d_in[16];
  const float* fc1w = (const float*)d_in[17];
  const float* fc1b = (const float*)d_in[18];
  const float* fc2w = (const float*)d_in[19];
  const float* fc2b = (const float*)d_in[20];

  float* ws    = (float*)d_ws;
  float* h1    = ws + OFF_H1;
  float* h2    = ws + OFF_H2;
  float* h3    = ws + OFF_H3;
  float* xz    = ws + OFF_XZ;
  float* u     = ws + OFF_U;
  float* xdbl  = ws + OFF_XDBL;
  float* delta = ws + OFF_DELTA;
  float* y     = ws + OFF_Y;
  float* seqA  = ws + OFF_SEQA;
  float* seqB  = ws + OFF_SEQB;

  // ---- conv stack ----
  conv1_kernel<<<4194304 / 256, 256, 0, stream>>>(x, c1w, c1b, b1g, b1b, h1);
  conv3x3_wmma<<<dim3(65536 / 32, 1), 128, 0, stream>>>(h1, c2w, c2b, b2g, b2b, h2,
                                                        64, 16, 32, 32, 64, 16, 32, 32, 1);
  conv3x3_wmma<<<dim3(8192 / 32, 2), 128, 0, stream>>>(h2, c3w, c3b, b3g, b3b, h3,
                                                       64, 16, 32, 32, 128, 8, 16, 16, 2);
  conv4_wmma<<<dim3(BLM / 32, 2), 128, 0, stream>>>(h3, c4w, c4b, b4g, b4b, seqA);

  // ---- two Mamba blocks ----
  auto run_block = [&](const float* seq_in, int base, float* seq_out) {
    const float* in_w = (const float*)d_in[base + 0];
    const float* cw   = (const float*)d_in[base + 1];
    const float* cbm  = (const float*)d_in[base + 2];
    const float* xpw  = (const float*)d_in[base + 3];
    const float* dtw  = (const float*)d_in[base + 4];
    const float* dtb  = (const float*)d_in[base + 5];
    const float* Alog = (const float*)d_in[base + 6];
    const float* Dp   = (const float*)d_in[base + 7];
    const float* outw = (const float*)d_in[base + 8];
    gemm_nt_wmma<<<dim3(BLM / 32, 8), 128, 0, stream>>>(seq_in, in_w, xz, BLM, 512, 128, 512);
    dwconv_silu_kernel<<<BLM, 256, 0, stream>>>(xz, cw, cbm, u);
    gemm_nt_wmma<<<dim3(BLM / 32, 1), 128, 0, stream>>>(u, xpw, xdbl, BLM, 40, 256, 40);
    delta_kernel<<<BLM, 256, 0, stream>>>(xdbl, dtw, dtb, delta);
    scan_kernel<<<dim3(DIN / 16, BATCH), 256, 0, stream>>>(xz, u, xdbl, delta, Alog, Dp, y);
    gemm_nt_wmma<<<dim3(BLM / 32, 2), 128, 0, stream>>>(y, outw, seq_out, BLM, 128, 256, 128);
  };
  run_block(seqA, 21, seqB);   // m1_*
  run_block(seqB, 30, seqA);   // m2_* (seqA is dead, reuse)

  // ---- head ----
  head_kernel<<<1, 256, 0, stream>>>(seqA, fc1w, fc1b, fc2w, fc2b, (float*)d_out);
}
